// Connection_36172214567767
// MI455X (gfx1250) — matrix operand order repaired and checked by reading, not compiled, not run on hardware
//
#include <hip/hip_runtime.h>
#include <hip/hip_bf16.h>
#include <math.h>

// ---------------------------------------------------------------------------
// Problem constants
// ---------------------------------------------------------------------------
#define Bq      8
#define L1q     1024
#define L2q     1024
#define DINq    1024
#define DOUTq   1024
#define Hq      16
#define DHq     64
#define KEYSq   2048            // L1 + L2
#define SCALEq  0.125f          // 1/sqrt(64)
#define NEG_INF (-__builtin_inff())

typedef __attribute__((ext_vector_type(16))) _Float16 v16h;
typedef __attribute__((ext_vector_type(8)))  _Float16 v8h;
typedef __attribute__((ext_vector_type(8)))  float    v8f;

// ---------------------------------------------------------------------------
// Workspace layout (bytes)
// ---------------------------------------------------------------------------
static constexpr size_t SZ_XK  = (size_t)Bq * KEYSq * DINq * 2;   // 33,554,432
static constexpr size_t SZ_W   = (size_t)DINq * DOUTq * 2;        //  2,097,152
static constexpr size_t SZ_QP  = (size_t)Bq * L2q * DOUTq * 2;    // 16,777,216
static constexpr size_t SZ_KP  = (size_t)Bq * KEYSq * DOUTq * 2;  // 33,554,432
static constexpr size_t SZ_VZ  = (size_t)Bq * Hq * 128 * 512 * 2; // 16,777,216

static constexpr size_t OFF_XK  = 0;
static constexpr size_t OFF_XV  = OFF_XK  + SZ_XK;
static constexpr size_t OFF_WQZ = OFF_XV  + SZ_XK;
static constexpr size_t OFF_WKZ = OFF_WQZ + SZ_W;
static constexpr size_t OFF_WVZ = OFF_WKZ + SZ_W;
static constexpr size_t OFF_WOZ = OFF_WVZ + SZ_W;
static constexpr size_t OFF_QP  = OFF_WOZ + SZ_W;
static constexpr size_t OFF_KP  = OFF_QP  + SZ_QP;
static constexpr size_t OFF_VP  = OFF_KP  + SZ_KP;
static constexpr size_t OFF_VZ  = OFF_VP  + SZ_KP;
static constexpr size_t OFF_ATT = OFF_VZ  + SZ_VZ;

// Attention LDS: 64 score rows, stride padded to keep 16B alignment for float4
#define SC_STRIDE 1032
static constexpr size_t SMEM_ATTN = (size_t)64 * SC_STRIDE * sizeof(float); // 264,192

// ---------------------------------------------------------------------------
// WMMA helpers (CDNA5 f16 -> f32, 16x16x32)
// ---------------------------------------------------------------------------
__device__ __forceinline__ v8f wmma16(v16h a, v16h b, v8f c) {
    return __builtin_amdgcn_wmma_f32_16x16x32_f16(
        /*neg_a=*/false, a, /*neg_b=*/false, b,
        /*c_mod=*/(short)0, c, /*reuse_a=*/false, /*reuse_b=*/false);
}

// A-fragment (16x32, f16) from a row-major [m, k] tile with row stride (halves).
// Contiguous-k layout used here (matches the B loads): lane L (m = L&15, kh = L>>4) holds K = kh*16 + 0..15
__device__ __forceinline__ v16h load_a_frag(const _Float16* base, int stride, int lane) {
    const int m  = lane & 15;
    const int kh = lane >> 4;
    const _Float16* p = base + (size_t)m * stride + kh * 16;
    v8h lo = *(const v8h*)(p);
    v8h hi = *(const v8h*)(p + 8);
    return __builtin_shufflevector(lo, hi, 0,1,2,3,4,5,6,7,8,9,10,11,12,13,14,15);
}

// B-fragment (32x16, f16) where the K-dimension is CONTIGUOUS in memory and the
// N-dimension has stride `stride` (halves): lane L (n = L&15, kh = L>>4) holds
// K = kh*16 + 0..15 as one 32-byte contiguous run.
__device__ __forceinline__ v16h load_b_kcontig(const _Float16* base, int stride, int lane) {
    const int n  = lane & 15;
    const int kh = lane >> 4;
    return *(const v16h*)(base + (size_t)n * stride + kh * 16);
}

// ---------------------------------------------------------------------------
// Prep kernels
// ---------------------------------------------------------------------------
// Xk = f16(concat(seq1 + b_prev, seq2 + b_self)); Xv = f16(concat(seq1, seq2))
__global__ __launch_bounds__(256) void prep_xkv(
    const float* __restrict__ seq1, const float* __restrict__ seq2,
    const float* __restrict__ bprev, const float* __restrict__ bself,
    _Float16* __restrict__ Xk, _Float16* __restrict__ Xv)
{
    size_t idx = (size_t)blockIdx.x * 256 + threadIdx.x;   // B*2048*1024 total
    int    d   = (int)(idx & 1023);
    size_t rl  = idx >> 10;
    int    l   = (int)(rl & 2047);
    size_t bb  = rl >> 11;
    float v, x;
    if (l < L1q) { v = seq1[(bb * L1q + l) * DINq + d];          x = v + bprev[d]; }
    else         { v = seq2[(bb * L2q + (l - L1q)) * DINq + d];  x = v + bself[d]; }
    Xk[idx] = (_Float16)x;
    Xv[idx] = (_Float16)v;
}

// Swizzle a 1024x1024 f32 weight into B-fragment order:
// Wz[(((kt*64)+nt)*32 + lane)*16 + e] = W[kt*32 + (lane>>4)*16 + e][nt*16 + (lane&15)]
__global__ __launch_bounds__(256) void prep_wswz(
    const float* __restrict__ W, _Float16* __restrict__ Wz)
{
    int idx  = blockIdx.x * 256 + threadIdx.x;    // 1,048,576 total
    int e    = idx & 15;
    int lane = (idx >> 4) & 31;
    int nt   = (idx >> 9) & 63;
    int kt   = idx >> 15;
    int n = nt * 16 + (lane & 15);
    int k = kt * 32 + ((lane >> 4) * 16) + e;
    Wz[idx] = (_Float16)W[(size_t)k * DOUTq + n];
}

// Swizzle the L1 part of V into B-fragment order with k-dim = key index:
// Vz[((b*16+h)*128 + kt*4 + nt)*512 + lane*16 + e]
//   = Vp[(b*2048 + kt*32 + (lane>>4)*16 + e)*1024 + h*64 + nt*16 + (lane&15)]
__global__ __launch_bounds__(256) void prep_vswz(
    const _Float16* __restrict__ Vp, _Float16* __restrict__ Vz)
{
    int idx  = blockIdx.x * 256 + threadIdx.x;    // 8,388,608 total
    int e    = idx & 15;
    int lane = (idx >> 4) & 31;
    int nt   = (idx >> 9) & 3;
    int kt   = (idx >> 11) & 31;
    int bh   = idx >> 16;
    int h = bh & 15, b = bh >> 4;
    int key = kt * 32 + (lane >> 4) * 16 + e;
    int dh  = nt * 16 + (lane & 15);
    Vz[idx] = Vp[((size_t)(b * KEYSq) + key) * DOUTq + h * DHq + dh];
}

// ---------------------------------------------------------------------------
// Generic GEMM: C[M,1024] = A[M,1024]_f16 @ Wswz + bias   (M multiple of 64)
// Row remap lets a logical-row space address strided batches inside A:
//   mem_row = (lrow / rpl) * mrpb + rstart + (lrow % rpl)
// ---------------------------------------------------------------------------
__global__ __launch_bounds__(256) void gemm_f16(
    const _Float16* __restrict__ A, const _Float16* __restrict__ Wz,
    const float* __restrict__ bias, _Float16* __restrict__ outH,
    float* __restrict__ outF, int rpl, int mrpb, int rstart)
{
    const int tid = threadIdx.x, w = tid >> 5, lane = tid & 31;
    const int wm = w & 3, wn = w >> 2;
    const int lrow0 = blockIdx.x * 64 + wm * 16;
    const int mem_row0 = (lrow0 / rpl) * mrpb + rstart + (lrow0 % rpl);
    const _Float16* abase = A + (size_t)mem_row0 * 1024;
    const int n0  = blockIdx.y * 64 + wn * 32;
    const int nt0 = n0 >> 4;

    v8f c0 = {}, c1 = {};
    for (int kt = 0; kt < 32; ++kt) {
        if (kt + 4 < 32) __builtin_prefetch(abase + (size_t)(kt + 4) * 32, 0, 0);
        v16h a = load_a_frag(abase + kt * 32, 1024, lane);
        const _Float16* wb = Wz + (((size_t)kt * 64 + nt0) * 32 + lane) * 16;
        v16h b0 = *(const v16h*)(wb);
        v16h b1 = *(const v16h*)(wb + 512);
        c0 = wmma16(a, b0, c0);
        c1 = wmma16(a, b1, c1);
    }

    v8f cc[2] = {c0, c1};
    for (int nt = 0; nt < 2; ++nt) {
        const int n = n0 + nt * 16 + (lane & 15);
        const float bv = bias[n];
        for (int r = 0; r < 8; ++r) {
            const int lr = lrow0 + r + 8 * (lane >> 4);
            const size_t o = (size_t)lr * 1024 + n;
            const float vv = cc[nt][r] + bv;
            if (outH) outH[o] = (_Float16)vv;
            if (outF) outF[o] = vv;
        }
    }
}

// ---------------------------------------------------------------------------
// Attention: one WG per (qt, h, b); 64 queries, full 1024-key score row in LDS
// ---------------------------------------------------------------------------
__device__ __forceinline__ v16h load_p_frag(const float* sc, int m0, int kbase, int lane) {
    const int m  = m0 + (lane & 15);
    const int kh = lane >> 4;
    const float* row = sc + (size_t)m * SC_STRIDE + kbase + kh * 16;
    float4 x0 = *(const float4*)(row);
    float4 x1 = *(const float4*)(row + 4);
    float4 y0 = *(const float4*)(row + 8);
    float4 y1 = *(const float4*)(row + 12);
    v16h p;
    p[0]=(_Float16)x0.x;  p[1]=(_Float16)x0.y;  p[2]=(_Float16)x0.z;  p[3]=(_Float16)x0.w;
    p[4]=(_Float16)x1.x;  p[5]=(_Float16)x1.y;  p[6]=(_Float16)x1.z;  p[7]=(_Float16)x1.w;
    p[8]=(_Float16)y0.x;  p[9]=(_Float16)y0.y;  p[10]=(_Float16)y0.z; p[11]=(_Float16)y0.w;
    p[12]=(_Float16)y1.x; p[13]=(_Float16)y1.y; p[14]=(_Float16)y1.z; p[15]=(_Float16)y1.w;
    return p;
}

__global__ __launch_bounds__(256) void attn_kernel(
    const _Float16* __restrict__ Qp, const _Float16* __restrict__ Kp,
    const _Float16* __restrict__ Vz, const _Float16* __restrict__ Vp,
    const int* __restrict__ seq_len1, _Float16* __restrict__ attn)
{
    extern __shared__ float sc[];           // [64][SC_STRIDE] scores
    __shared__ float s_pmax[4][64];
    __shared__ float s_psum[4][64];
    __shared__ float s_red[64];             // rowmax, then 1/sum
    __shared__ float s_sself[64];
    __shared__ float s_wself[64];

    const int qt = blockIdx.x, h = blockIdx.y, b = blockIdx.z;
    const int q0 = qt * 64;
    const int tid = threadIdx.x, w = tid >> 5, lane = tid & 31;
    const int wm = w & 3, wn = w >> 2;
    const int len = seq_len1[b];

    // ---- diagonal self logits -------------------------------------------
    if (tid < 64) {
        const int m = tid, row = q0 + m;
        const _Float16* qp = Qp + ((size_t)(b * L2q + row)) * DOUTq + h * DHq;
        const _Float16* kp = Kp + ((size_t)(b * KEYSq + L1q + row)) * DOUTq + h * DHq;
        float s = 0.f;
        for (int d = 0; d < DHq; ++d) s += (float)qp[d] * (float)kp[d];
        s_sself[m] = s * SCALEq;
    }

    // ---- S = Q.K^T over L1 keys (WMMA), scaled + masked, into LDS -------
    {
        const int m0 = wm * 16;
        const _Float16* qbase = Qp + ((size_t)(b * L2q + q0 + m0)) * DOUTq + h * DHq;
        v16h qa0 = load_a_frag(qbase,       DOUTq, lane);   // dh 0..31
        v16h qa1 = load_a_frag(qbase + 32,  DOUTq, lane);   // dh 32..63
        for (int kc = wn; kc < 64; kc += 2) {               // 16-key chunks
            const _Float16* kb = Kp + ((size_t)(b * KEYSq + kc * 16)) * DOUTq + h * DHq;
            v16h kb0 = load_b_kcontig(kb,      DOUTq, lane);
            v16h kb1 = load_b_kcontig(kb + 32, DOUTq, lane);
            v8f c = {};
            c = wmma16(qa0, kb0, c);
            c = wmma16(qa1, kb1, c);
            const int key = kc * 16 + (lane & 15);
            const bool ok = key < len;
            for (int r = 0; r < 8; ++r) {
                const int m = m0 + r + 8 * (lane >> 4);
                sc[(size_t)m * SC_STRIDE + key] = ok ? c[r] * SCALEq : NEG_INF;
            }
        }
    }
    __syncthreads();

    // ---- masked softmax over (L1 keys + self), normalize in LDS ---------
    {
        const int m = tid & 63, part = tid >> 6;
        float* rowp = sc + (size_t)m * SC_STRIDE;
        const int kb = part * 256;
        float mx = NEG_INF;
        for (int k = 0; k < 256; ++k) mx = fmaxf(mx, rowp[kb + k]);
        if (part == 0) mx = fmaxf(mx, s_sself[m]);
        s_pmax[part][m] = mx;
        __syncthreads();
        if (tid < 64) {
            s_red[tid] = fmaxf(fmaxf(s_pmax[0][tid], s_pmax[1][tid]),
                               fmaxf(s_pmax[2][tid], s_pmax[3][tid]));
        }
        __syncthreads();
        const float rm = s_red[m];
        float sum = 0.f;
        for (int k = 0; k < 256; ++k) {
            float e = __expf(rowp[kb + k] - rm);
            rowp[kb + k] = e;
            sum += e;
        }
        if (part == 0) sum += __expf(s_sself[m] - rm);
        s_psum[part][m] = sum;
        __syncthreads();
        if (tid < 64) {
            const float tot = s_psum[0][tid] + s_psum[1][tid] +
                              s_psum[2][tid] + s_psum[3][tid];
            const float inv = 1.0f / tot;
            s_wself[tid] = __expf(s_sself[tid] - s_red[tid]) * inv;
            s_red[tid]   = inv;
        }
        __syncthreads();
        const float inv = s_red[m];
        for (int k = 0; k < 256; ++k) rowp[kb + k] *= inv;
    }
    __syncthreads();

    // ---- attn = P.V (WMMA over pre-swizzled V) + w_self * v_self --------
    {
        const int m0 = wm * 16;
        v8f acc0 = {}, acc1 = {};
        const _Float16* vzb = Vz + ((size_t)(b * Hq + h)) * 128 * 512;
        for (int kc = 0; kc < 32; ++kc) {                   // 32-key chunks
            v16h pa = load_p_frag(sc, m0, kc * 32, lane);
            const _Float16* vb = vzb + ((size_t)kc * 4 + wn * 2) * 512 + lane * 16;
            v16h vb0 = *(const v16h*)(vb);
            v16h vb1 = *(const v16h*)(vb + 512);
            acc0 = wmma16(pa, vb0, acc0);
            acc1 = wmma16(pa, vb1, acc1);
        }
        v8f cc[2] = {acc0, acc1};
        for (int nt = 0; nt < 2; ++nt) {
            const int n = wn * 32 + nt * 16 + (lane & 15);
            for (int r = 0; r < 8; ++r) {
                const int m = m0 + r + 8 * (lane >> 4);
                const int qrow = q0 + m;
                const float vself =
                    (float)Vp[((size_t)(b * KEYSq + L1q + qrow)) * DOUTq + h * DHq + n];
                const float val = cc[nt][r] + s_wself[m] * vself;
                attn[((size_t)(b * L2q + qrow)) * DOUTq + h * DHq + n] = (_Float16)val;
            }
        }
    }
}

// ---------------------------------------------------------------------------
// Host launcher
// ---------------------------------------------------------------------------
extern "C" void kernel_launch(void* const* d_in, const int* in_sizes, int n_in,
                              void* d_out, int out_size, void* d_ws, size_t ws_size,
                              hipStream_t stream) {
    (void)in_sizes; (void)n_in; (void)out_size; (void)ws_size;
    const float* seq1  = (const float*)d_in[0];
    const float* seq2  = (const float*)d_in[1];
    const int*   slen  = (const int*)  d_in[2];
    const float* Wq    = (const float*)d_in[3];
    const float* bq    = (const float*)d_in[4];
    const float* Wk    = (const float*)d_in[5];
    const float* bk    = (const float*)d_in[6];
    const float* Wv    = (const float*)d_in[7];
    const float* bv    = (const float*)d_in[8];
    const float* Wo    = (const float*)d_in[9];
    const float* bo    = (const float*)d_in[10];
    const float* bprev = (const float*)d_in[11];
    const float* bself = (const float*)d_in[12];
    float* out = (float*)d_out;

    char* ws = (char*)d_ws;
    _Float16* Xk  = (_Float16*)(ws + OFF_XK);
    _Float16* Xv  = (_Float16*)(ws + OFF_XV);
    _Float16* Wqz = (_Float16*)(ws + OFF_WQZ);
    _Float16* Wkz = (_Float16*)(ws + OFF_WKZ);
    _Float16* Wvz = (_Float16*)(ws + OFF_WVZ);
    _Float16* Woz = (_Float16*)(ws + OFF_WOZ);
    _Float16* Qp  = (_Float16*)(ws + OFF_QP);
    _Float16* Kp  = (_Float16*)(ws + OFF_KP);
    _Float16* Vp  = (_Float16*)(ws + OFF_VP);
    _Float16* Vz  = (_Float16*)(ws + OFF_VZ);
    _Float16* Att = (_Float16*)(ws + OFF_ATT);

    // 1) biased f16 inputs + f16 weight swizzles
    prep_xkv<<<65536, 256, 0, stream>>>(seq1, seq2, bprev, bself, Xk, Xv);
    prep_wswz<<<4096, 256, 0, stream>>>(Wq, Wqz);
    prep_wswz<<<4096, 256, 0, stream>>>(Wk, Wkz);
    prep_wswz<<<4096, 256, 0, stream>>>(Wv, Wvz);
    prep_wswz<<<4096, 256, 0, stream>>>(Wo, Woz);

    // 2) projections (WMMA GEMMs) — query rows alias Xk's second half
    gemm_f16<<<dim3(128, 16), 256, 0, stream>>>(Xk, Wqz, bq, Qp, nullptr, 1024, 2048, 1024);
    gemm_f16<<<dim3(256, 16), 256, 0, stream>>>(Xk, Wkz, bk, Kp, nullptr, 2048, 2048, 0);
    gemm_f16<<<dim3(256, 16), 256, 0, stream>>>(Xv, Wvz, bv, Vp, nullptr, 2048, 2048, 0);

    // 3) swizzle V (L1 part) for the P.V GEMM
    prep_vswz<<<32768, 256, 0, stream>>>(Vp, Vz);

    // 4) attention with full-row LDS softmax (264 KB dynamic LDS per WG)
    hipFuncSetAttribute(reinterpret_cast<const void*>(attn_kernel),
                        hipFuncAttributeMaxDynamicSharedMemorySize, (int)SMEM_ATTN);
    attn_kernel<<<dim3(16, 16, 8), 256, SMEM_ATTN, stream>>>(Qp, Kp, Vz, Vp, slen, Att);

    // 5) output projection -> f32 d_out
    gemm_f16<<<dim3(128, 16), 256, 0, stream>>>(Att, Woz, bo, nullptr, out, 1024, 1024, 0);
}
